// Decoder_57913339020067
// MI455X (gfx1250) — matrix operand order repaired and checked by reading, not compiled, not run on hardware
//
#include <hip/hip_runtime.h>
#include <hip/hip_bf16.h>
#include <math.h>

// Problem dims (fixed by the reference)
#define B_   64
#define P_   196      // 14*14 pixels
#define ENC_ 2048
#define E_   512
#define D_   512
#define A_   512
#define V_   32000
#define L_   32
#define T_   31       // L-1 decode steps
#define XW_  (E_ + ENC_)   // 2560
#define LN_EPS 1e-5f

typedef __attribute__((ext_vector_type(16))) __bf16        v16bf;
typedef __attribute__((ext_vector_type(8)))  float         v8f;
typedef __attribute__((ext_vector_type(4)))  unsigned int  u32x4;

static __device__ __forceinline__ float sigm(float x) { return 1.0f / (1.0f + __expf(-x)); }

union BFV { v16bf v; u32x4 q[2]; };

// ---------------------------------------------------------------------------
// M=64 GEMM: one wave owns ALL 4 M-tiles (4 x v8f accumulators) for one N-tile,
// so each B (weight) panel is loaded exactly once. Per K-step:
// 2x B global_load_b128 + 8x A global_load_b128 + 4x v_wmma.
// C[m][n] = sum_k A[m][k]*W[n][k] + bias[n]  (+C if ACC) (time-masked if MASK)
// ---------------------------------------------------------------------------
template<bool ACC, bool MASK>
__global__ __launch_bounds__(128) void gemm_mt4(
    const __bf16* __restrict__ A, int lda,   // 64 x K
    const __bf16* __restrict__ W, int ldw,   // N x K
    const float* __restrict__ bias,
    float* __restrict__ C, int ldc,
    int N, int K,
    const int* __restrict__ declen, int t)
{
  const int wave  = threadIdx.x >> 5;
  const int lane  = threadIdx.x & 31;
  const int ntile = blockIdx.x * 4 + wave;
  if (ntile * 16 >= N) return;
  const int half = lane >> 4;   // 0: lanes 0-15, 1: lanes 16-31
  const int r    = lane & 15;

  const __bf16* Ap = A + (size_t)r * lda + half * 8;                  // + mt*16*lda
  const __bf16* Wp = W + (size_t)(ntile * 16 + r) * ldw + half * 8;

  v8f acc[4];
  #pragma unroll
  for (int mt = 0; mt < 4; ++mt) acc[mt] = (v8f){0.f,0.f,0.f,0.f,0.f,0.f,0.f,0.f};

  for (int k0 = 0; k0 < K; k0 += 32) {
    BFV b;
    b.q[0] = *(const u32x4*)(Wp + k0);        // K = k0 + half*8 + [0..7]
    b.q[1] = *(const u32x4*)(Wp + k0 + 16);   // K = k0 + 16 + half*8 + [0..7]
    __builtin_prefetch(Wp + k0 + 128, 0, 3);  // speculative, next W chunk
    #pragma unroll
    for (int mt = 0; mt < 4; ++mt) {
      BFV a;
      const __bf16* ap = Ap + (size_t)mt * 16 * lda;
      a.q[0] = *(const u32x4*)(ap + k0);      // K = k0 + half*8 + [0..7]
      a.q[1] = *(const u32x4*)(ap + k0 + 16); // K = k0 + 16 + half*8 + [0..7]
      acc[mt] = __builtin_amdgcn_wmma_f32_16x16x32_bf16(
          false, a.v, false, b.v, (short)0, acc[mt], false, false);
    }
  }

  // C layout: vgpr v, lane: M = v + half*8, N = r
  const int n = ntile * 16 + r;
  const float bv = bias[n];
  #pragma unroll
  for (int mt = 0; mt < 4; ++mt) {
    const int mbase = mt * 16 + half * 8;
    #pragma unroll
    for (int v = 0; v < 8; ++v) {
      const long idx = (long)(mbase + v) * ldc + n;
      float val = acc[mt][v] + bv;
      if (ACC) val += C[idx];
      if (MASK) val = (t < declen[mbase + v]) ? val : 0.0f;
      C[idx] = val;
    }
  }
}

// ---------------------------------------------------------------------------
// att1 GEMM (M=12544, N=512, K=2048): the 4 waves of a block share one 16-row
// A panel (16x2048 bf16 = 64 KB, contiguous). Stage it once per block into LDS
// with global_load_async_to_lds_b128 (ASYNCcnt), then K-loop reads A from LDS
// (ds_load_b128) and B from global (global_load_b128).
// ---------------------------------------------------------------------------
__global__ __launch_bounds__(128) void gemm_att1_lds(
    const __bf16* __restrict__ A,   // 12544 x 2048 (permuted enc, bf16)
    const __bf16* __restrict__ W,   // 512 x 2048
    const float* __restrict__ bias,
    float* __restrict__ C)          // 12544 x 512
{
  __shared__ __bf16 sA[16 * ENC_];  // 64 KB
  const int wave  = threadIdx.x >> 5;
  const int lane  = threadIdx.x & 31;
  const int ntile = blockIdx.x * 4 + wave;
  const int mtile = blockIdx.y;

  // async-stage A panel: 64 KB = 128 threads x 32 x 16 B
  {
    const char* gbase = (const char*)(A + (size_t)mtile * 16 * ENC_);
    const unsigned lbase = (unsigned)(size_t)(&sA[0]);
    #pragma unroll 4
    for (int ofs = threadIdx.x * 16; ofs < 16 * ENC_ * 2; ofs += 128 * 16) {
      const unsigned laddr = lbase + ofs;
      const char* gaddr = gbase + ofs;
      asm volatile("global_load_async_to_lds_b128 %0, %1, off"
                   :: "v"(laddr), "v"(gaddr) : "memory");
    }
    asm volatile("s_wait_asynccnt 0" ::: "memory");
    __syncthreads();
  }

  const int half = lane >> 4;
  const int r    = lane & 15;
  const __bf16* sAp = sA + r * ENC_ + half * 8;
  const __bf16* Wp  = W + (size_t)(ntile * 16 + r) * ENC_ + half * 8;

  v8f acc = {0.f, 0.f, 0.f, 0.f, 0.f, 0.f, 0.f, 0.f};
  #pragma unroll 2
  for (int k0 = 0; k0 < ENC_; k0 += 32) {
    BFV a, b;
    a.q[0] = *(const u32x4*)(sAp + k0);        // ds_load_b128
    a.q[1] = *(const u32x4*)(sAp + k0 + 16);
    b.q[0] = *(const u32x4*)(Wp + k0);
    b.q[1] = *(const u32x4*)(Wp + k0 + 16);
    acc = __builtin_amdgcn_wmma_f32_16x16x32_bf16(
        false, a.v, false, b.v, (short)0, acc, false, false);
  }

  const int n = ntile * 16 + r;
  const float bv = bias[n];
  const int mbase = mtile * 16 + half * 8;
  #pragma unroll
  for (int v = 0; v < 8; ++v)
    C[(long)(mbase + v) * A_ + n] = acc[v] + bv;
}

// f32 -> bf16 streaming conversion (weights, done once per call)
__global__ void cvt_bf16_kernel(const float* __restrict__ src, __bf16* __restrict__ dst, long n)
{
  long i = (long)blockIdx.x * blockDim.x + threadIdx.x;
  const long stride = (long)gridDim.x * blockDim.x;
  for (; i < n; i += stride) dst[i] = (__bf16)src[i];
}

// Permuted encoder rows -> bf16 (applies batch sort; att1 GEMM then needs no gather)
__global__ void cvt_enc_kernel(const float* __restrict__ enc, const int* __restrict__ rowmap,
                               __bf16* __restrict__ dst)
{
  const int rrow = blockIdx.x;
  const float* src = enc + (long)rowmap[rrow] * ENC_;
  __bf16* d = dst + (long)rrow * ENC_;
  for (int j = threadIdx.x; j < ENC_; j += 256) d[j] = (__bf16)src[j];
}

// ---------------------------------------------------------------------------
// Stable descending argsort of lengths (B=64), reorder captions, build row map,
// emit caps/decode_lengths/order outputs as f32.
// ---------------------------------------------------------------------------
__global__ void sort_kernel(const int* __restrict__ lengths, const int* __restrict__ caps,
                            int* __restrict__ order, int* __restrict__ declen,
                            int* __restrict__ caps_sorted, int* __restrict__ rowmap,
                            float* __restrict__ out_caps, float* __restrict__ out_dlen,
                            float* __restrict__ out_order)
{
  __shared__ int s_len[B_];
  __shared__ int s_ord[B_];
  const int tid = threadIdx.x;
  if (tid < B_) s_len[tid] = lengths[tid];
  __syncthreads();
  if (tid < B_) {
    const int len = s_len[tid];
    int rank = 0;
    for (int j = 0; j < B_; ++j) {
      const int lj = s_len[j];
      rank += (lj > len) || (lj == len && j < tid);   // stable, descending
    }
    s_ord[rank] = tid;
  }
  __syncthreads();
  if (tid < B_) {
    const int o = s_ord[tid];
    order[tid] = o;
    const int dl = s_len[o] - 1;
    declen[tid]    = dl;
    out_dlen[tid]  = (float)dl;
    out_order[tid] = (float)o;
    for (int k = 0; k < L_; ++k) {
      const int cv = caps[o * L_ + k];
      caps_sorted[tid * L_ + k] = cv;
      out_caps[tid * L_ + k] = (float)cv;
    }
    for (int p = 0; p < P_; ++p) rowmap[tid * P_ + p] = o * P_ + p;
  }
}

// mean over 196 pixels of the permuted encoder output -> bf16 (GEMM A operand)
__global__ void mean_kernel(const float* __restrict__ enc, const int* __restrict__ order,
                            __bf16* __restrict__ mean_bf)
{
  const int b = blockIdx.x;
  const int e = blockIdx.y * blockDim.x + threadIdx.x;
  const float* base = enc + (long)order[b] * P_ * ENC_ + e;
  float s = 0.f;
  for (int p = 0; p < P_; ++p) s += base[(long)p * ENC_];
  mean_bf[b * ENC_ + e] = (__bf16)(s * (1.0f / P_));
}

// LayerNorm over D_=512, one block per row; OUT_BF selects bf16 vs f32 output
template<bool OUT_BF>
__global__ void ln_kernel(const float* __restrict__ x, const float* __restrict__ g,
                          const float* __restrict__ be,
                          float* __restrict__ outf, __bf16* __restrict__ outb)
{
  __shared__ float s1[256], s2[256];
  const int b = blockIdx.x, tid = threadIdx.x;
  const float v0 = x[b * D_ + tid], v1 = x[b * D_ + tid + 256];
  s1[tid] = v0 + v1; s2[tid] = v0 * v0 + v1 * v1;
  __syncthreads();
  for (int st = 128; st > 0; st >>= 1) {
    if (tid < st) { s1[tid] += s1[tid + st]; s2[tid] += s2[tid + st]; }
    __syncthreads();
  }
  const float m   = s1[0] / D_;
  const float var = s2[0] / D_ - m * m;
  const float inv = rsqrtf(var + LN_EPS);
  const float o0 = (v0 - m) * inv * g[tid]       + be[tid];
  const float o1 = (v1 - m) * inv * g[tid + 256] + be[tid + 256];
  if (OUT_BF) {
    outb[b * D_ + tid]       = (__bf16)o0;
    outb[b * D_ + tid + 256] = (__bf16)o1;
  } else {
    outf[b * D_ + tid]       = o0;
    outf[b * D_ + tid + 256] = o1;
  }
}

// e = relu(att1 + att2)·w_full + b_full ; softmax over 196 ; masked alpha out
__global__ void att_score_kernel(const float* __restrict__ att1, const float* __restrict__ att2,
                                 const float* __restrict__ wfull, const float* __restrict__ bfull,
                                 float* __restrict__ alpha_ws, float* __restrict__ alpha_out,
                                 int alpha_stride, const int* __restrict__ declen, int t)
{
  __shared__ float s_att2[A_], s_wf[A_], s_red[256];
  const int b = blockIdx.x, tid = threadIdx.x;
  s_att2[tid]       = att2[b * A_ + tid];
  s_att2[tid + 256] = att2[b * A_ + tid + 256];
  s_wf[tid]       = wfull[tid];
  s_wf[tid + 256] = wfull[tid + 256];
  __syncthreads();
  float e = -1e30f;
  if (tid < P_) {
    const float* a1 = att1 + ((long)b * P_ + tid) * A_;
    float acc = bfull[0];
    for (int a = 0; a < A_; ++a) {
      const float s = a1[a] + s_att2[a];
      acc += fmaxf(s, 0.f) * s_wf[a];
    }
    e = acc;
  }
  s_red[tid] = e; __syncthreads();
  for (int st = 128; st > 0; st >>= 1) {
    if (tid < st) s_red[tid] = fmaxf(s_red[tid], s_red[tid + st]);
    __syncthreads();
  }
  const float mx = s_red[0]; __syncthreads();
  const float ex = (tid < P_) ? expf(e - mx) : 0.f;
  s_red[tid] = ex; __syncthreads();
  for (int st = 128; st > 0; st >>= 1) {
    if (tid < st) s_red[tid] += s_red[tid + st];
    __syncthreads();
  }
  const float inv = 1.0f / s_red[0];
  if (tid < P_) {
    const float al = ex * inv;
    alpha_ws[b * P_ + tid] = al;
    alpha_out[(long)b * alpha_stride + tid] = (t < declen[b]) ? al : 0.f;
  }
}

// awe[b][e] = sum_p alpha[b][p] * enc_bf[b][p][e]  (bf16 permuted enc, L2-resident)
__global__ void awe_kernel(const __bf16* __restrict__ enc_bf,
                           const float* __restrict__ alpha, float* __restrict__ awe)
{
  __shared__ float s_al[P_];
  const int b = blockIdx.x, tid = threadIdx.x;
  if (tid < P_) s_al[tid] = alpha[b * P_ + tid];
  __syncthreads();
  const int e = blockIdx.y * blockDim.x + tid;
  const __bf16* base = enc_bf + (long)b * P_ * ENC_ + e;
  float s = 0.f;
  for (int p = 0; p < P_; ++p) s += (float)base[(long)p * ENC_] * s_al[p];
  awe[b * ENC_ + e] = s;
}

// x_bf = [ emb[cap_t] , sigmoid(gatepre) * awe ]  written directly as bf16
__global__ void build_x_kernel(const float* __restrict__ emb, const int* __restrict__ caps_sorted,
                               const float* __restrict__ gatepre, const float* __restrict__ awe,
                               __bf16* __restrict__ x, int t)
{
  const int b = blockIdx.x, tid = threadIdx.x;
  const int cap = caps_sorted[b * L_ + t];
  const float* erow = emb + (long)cap * E_;
  for (int j = tid; j < E_; j += 256) x[b * XW_ + j] = (__bf16)erow[j];
  for (int j = tid; j < ENC_; j += 256) {
    const float gp = gatepre[b * ENC_ + j];
    x[b * XW_ + E_ + j] = (__bf16)(sigm(gp) * awe[b * ENC_ + j]);
  }
}

// LSTM pointwise + LayerNorm(h_new, c_new); hnew_bf always written (for FC GEMM),
// h_bf / c updated only when t < decode_len[b]
__global__ void lstm_kernel(const float* __restrict__ gates, float* __restrict__ c,
                            __bf16* __restrict__ h_bf, __bf16* __restrict__ hnew_bf,
                            const float* __restrict__ g_h, const float* __restrict__ be_h,
                            const float* __restrict__ g_c, const float* __restrict__ be_c,
                            const int* __restrict__ declen, int t)
{
  __shared__ float sh1[256], sh2[256], sc1[256], sc2[256];
  const int b = blockIdx.x, tid = threadIdx.x;
  const float* gr = gates + (long)b * (4 * D_);
  float hn[2], cn[2];
  #pragma unroll
  for (int q = 0; q < 2; ++q) {
    const int ch = tid + q * 256;
    const float iv = sigm(gr[ch]);
    const float fv = sigm(gr[D_ + ch]);
    const float gv = tanhf(gr[2 * D_ + ch]);
    const float ov = sigm(gr[3 * D_ + ch]);
    const float cv = fv * c[b * D_ + ch] + iv * gv;
    cn[q] = cv;
    hn[q] = ov * tanhf(cv);
  }
  sh1[tid] = hn[0] + hn[1]; sh2[tid] = hn[0] * hn[0] + hn[1] * hn[1];
  sc1[tid] = cn[0] + cn[1]; sc2[tid] = cn[0] * cn[0] + cn[1] * cn[1];
  __syncthreads();
  for (int st = 128; st > 0; st >>= 1) {
    if (tid < st) {
      sh1[tid] += sh1[tid + st]; sh2[tid] += sh2[tid + st];
      sc1[tid] += sc1[tid + st]; sc2[tid] += sc2[tid + st];
    }
    __syncthreads();
  }
  const float mh = sh1[0] / D_, vh = sh2[0] / D_ - mh * mh;
  const float mc = sc1[0] / D_, vc = sc2[0] / D_ - mc * mc;
  const float ih = rsqrtf(vh + LN_EPS), ic = rsqrtf(vc + LN_EPS);
  const int mask = t < declen[b];
  #pragma unroll
  for (int q = 0; q < 2; ++q) {
    const int ch = tid + q * 256;
    const float hl = (hn[q] - mh) * ih * g_h[ch] + be_h[ch];
    const float cl = (cn[q] - mc) * ic * g_c[ch] + be_c[ch];
    hnew_bf[b * D_ + ch] = (__bf16)hl;       // preds always use h_new
    if (mask) { h_bf[b * D_ + ch] = (__bf16)hl; c[b * D_ + ch] = cl; }
  }
}

// ---------------------------------------------------------------------------
extern "C" void kernel_launch(void* const* d_in, const int* in_sizes, int n_in,
                              void* d_out, int out_size, void* d_ws, size_t ws_size,
                              hipStream_t stream)
{
  (void)in_sizes; (void)n_in; (void)out_size; (void)ws_size;
  const float* enc       = (const float*)d_in[0];
  const int*   caps      = (const int*)  d_in[1];
  const int*   lengths   = (const int*)  d_in[2];
  const float* emb       = (const float*)d_in[3];
  const float* W_enc_att = (const float*)d_in[4];
  const float* b_enc_att = (const float*)d_in[5];
  const float* W_dec_att = (const float*)d_in[6];
  const float* b_dec_att = (const float*)d_in[7];
  const float* W_full    = (const float*)d_in[8];
  const float* b_full    = (const float*)d_in[9];
  const float* W_ih      = (const float*)d_in[10];
  const float* b_ih      = (const float*)d_in[11];
  const float* W_hh      = (const float*)d_in[12];
  const float* b_hh      = (const float*)d_in[13];
  const float* g_h       = (const float*)d_in[14];
  const float* be_h      = (const float*)d_in[15];
  const float* g_c       = (const float*)d_in[16];
  const float* be_c      = (const float*)d_in[17];
  const float* W_init_h  = (const float*)d_in[18];
  const float* b_init_h  = (const float*)d_in[19];
  const float* W_init_c  = (const float*)d_in[20];
  const float* b_init_c  = (const float*)d_in[21];
  const float* W_beta    = (const float*)d_in[22];
  const float* b_beta    = (const float*)d_in[23];
  const float* W_fc      = (const float*)d_in[24];
  const float* b_fc      = (const float*)d_in[25];

  // Output layout (all f32): preds(B,T,V) | caps(B,L) | declen(B) | alphas(B,T,P) | order(B)
  float* out       = (float*)d_out;
  float* out_preds = out;
  float* out_caps  = out_preds + (size_t)B_ * T_ * V_;
  float* out_dlen  = out_caps  + (size_t)B_ * L_;
  float* out_alpha = out_dlen  + B_;
  float* out_order = out_alpha + (size_t)B_ * T_ * P_;

  // Workspace carve-up (~134 MB)
  char* w = (char*)d_ws;
  auto alloc = [&](size_t bytes) { void* p = (void*)w; w += (bytes + 255) & ~(size_t)255; return p; };
  int*   order   = (int*)  alloc(B_ * 4);
  int*   declen  = (int*)  alloc(B_ * 4);
  int*   caps_s  = (int*)  alloc((size_t)B_ * L_ * 4);
  int*   rowmap  = (int*)  alloc((size_t)B_ * P_ * 4);
  float* att1    = (float*)alloc((size_t)B_ * P_ * A_ * 4);
  float* att2    = (float*)alloc((size_t)B_ * A_ * 4);
  float* alpha   = (float*)alloc((size_t)B_ * P_ * 4);
  float* awe     = (float*)alloc((size_t)B_ * ENC_ * 4);
  float* gatepre = (float*)alloc((size_t)B_ * ENC_ * 4);
  float* gates   = (float*)alloc((size_t)B_ * 4 * D_ * 4);
  float* cst     = (float*)alloc((size_t)B_ * D_ * 4);
  float* pre_h   = (float*)alloc((size_t)B_ * D_ * 4);
  float* pre_c   = (float*)alloc((size_t)B_ * D_ * 4);
  __bf16* enc_bf  = (__bf16*)alloc((size_t)B_ * P_ * ENC_ * 2);  // permuted
  __bf16* mean_bf = (__bf16*)alloc((size_t)B_ * ENC_ * 2);
  __bf16* h_bf    = (__bf16*)alloc((size_t)B_ * D_ * 2);
  __bf16* hnew_bf = (__bf16*)alloc((size_t)B_ * D_ * 2);
  __bf16* x_bf    = (__bf16*)alloc((size_t)B_ * XW_ * 2);
  __bf16* Wenc_bf = (__bf16*)alloc((size_t)A_ * ENC_ * 2);
  __bf16* Wdec_bf = (__bf16*)alloc((size_t)A_ * D_ * 2);
  __bf16* Wbeta_bf= (__bf16*)alloc((size_t)ENC_ * D_ * 2);
  __bf16* Wih_bf  = (__bf16*)alloc((size_t)4 * D_ * XW_ * 2);
  __bf16* Whh_bf  = (__bf16*)alloc((size_t)4 * D_ * D_ * 2);
  __bf16* Wfc_bf  = (__bf16*)alloc((size_t)V_ * D_ * 2);
  __bf16* Wih0_bf = (__bf16*)alloc((size_t)D_ * ENC_ * 2);   // W_init_h
  __bf16* Wic0_bf = (__bf16*)alloc((size_t)D_ * ENC_ * 2);   // W_init_c

  // ---- setup ----
  sort_kernel<<<1, 64, 0, stream>>>(lengths, caps, order, declen, caps_s, rowmap,
                                    out_caps, out_dlen, out_order);
  cvt_enc_kernel<<<B_ * P_, 256, 0, stream>>>(enc, rowmap, enc_bf);
  cvt_bf16_kernel<<<2048, 256, 0, stream>>>(W_enc_att, Wenc_bf, (long)A_ * ENC_);
  cvt_bf16_kernel<<<1024, 256, 0, stream>>>(W_dec_att, Wdec_bf, (long)A_ * D_);
  cvt_bf16_kernel<<<2048, 256, 0, stream>>>(W_beta,    Wbeta_bf,(long)ENC_ * D_);
  cvt_bf16_kernel<<<4096, 256, 0, stream>>>(W_ih,      Wih_bf,  (long)4 * D_ * XW_);
  cvt_bf16_kernel<<<2048, 256, 0, stream>>>(W_hh,      Whh_bf,  (long)4 * D_ * D_);
  cvt_bf16_kernel<<<8192, 256, 0, stream>>>(W_fc,      Wfc_bf,  (long)V_ * D_);
  cvt_bf16_kernel<<<2048, 256, 0, stream>>>(W_init_h,  Wih0_bf, (long)D_ * ENC_);
  cvt_bf16_kernel<<<2048, 256, 0, stream>>>(W_init_c,  Wic0_bf, (long)D_ * ENC_);
  mean_kernel<<<dim3(B_, ENC_ / 256), 256, 0, stream>>>(enc, order, mean_bf);

  // att1 = enc_sorted @ W_enc_att^T + b   (LDS-staged A panel, async loads)
  gemm_att1_lds<<<dim3(A_ / 64, (B_ * P_) / 16), 128, 0, stream>>>(
      enc_bf, Wenc_bf, b_enc_att, att1);

  // h,c init: LN(mean @ W_init^T + b)
  gemm_mt4<false, false><<<dim3(D_ / 64), 128, 0, stream>>>(
      mean_bf, ENC_, Wih0_bf, ENC_, b_init_h, pre_h, D_, D_, ENC_, nullptr, 0);
  gemm_mt4<false, false><<<dim3(D_ / 64), 128, 0, stream>>>(
      mean_bf, ENC_, Wic0_bf, ENC_, b_init_c, pre_c, D_, D_, ENC_, nullptr, 0);
  ln_kernel<true ><<<B_, 256, 0, stream>>>(pre_h, g_h, be_h, nullptr, h_bf);
  ln_kernel<false><<<B_, 256, 0, stream>>>(pre_c, g_c, be_c, cst, nullptr);

  // ---- 31 decode steps ----
  for (int t = 0; t < T_; ++t) {
    // att2 = h @ W_dec_att^T + b
    gemm_mt4<false, false><<<dim3(A_ / 64), 128, 0, stream>>>(
        h_bf, D_, Wdec_bf, D_, b_dec_att, att2, A_, A_, D_, nullptr, 0);
    att_score_kernel<<<B_, 256, 0, stream>>>(att1, att2, W_full, b_full,
        alpha, out_alpha + (size_t)t * P_, T_ * P_, declen, t);
    awe_kernel<<<dim3(B_, ENC_ / 256), 256, 0, stream>>>(enc_bf, alpha, awe);
    // beta gate pre-activation = h @ W_beta^T + b
    gemm_mt4<false, false><<<dim3(ENC_ / 64), 128, 0, stream>>>(
        h_bf, D_, Wbeta_bf, D_, b_beta, gatepre, ENC_, ENC_, D_, nullptr, 0);
    build_x_kernel<<<B_, 256, 0, stream>>>(emb, caps_s, gatepre, awe, x_bf, t);
    // gates = x @ W_ih^T + b_ih + h @ W_hh^T + b_hh
    gemm_mt4<false, false><<<dim3((4 * D_) / 64), 128, 0, stream>>>(
        x_bf, XW_, Wih_bf, XW_, b_ih, gates, 4 * D_, 4 * D_, XW_, nullptr, 0);
    gemm_mt4<true, false><<<dim3((4 * D_) / 64), 128, 0, stream>>>(
        h_bf, D_, Whh_bf, D_, b_hh, gates, 4 * D_, 4 * D_, D_, nullptr, 0);
    lstm_kernel<<<B_, 256, 0, stream>>>(gates, cst, h_bf, hnew_bf,
                                        g_h, be_h, g_c, be_c, declen, t);
    // preds -> out[b][t][v], masked; C base offset t*V, row stride T*V
    gemm_mt4<false, true><<<dim3(V_ / 64), 128, 0, stream>>>(
        hnew_bf, D_, Wfc_bf, D_, b_fc, out_preds + (size_t)t * V_, T_ * V_,
        V_, D_, declen, t);
  }
}
